// get_gen_model_21715354648782
// MI455X (gfx1250) — compile-verified, matrix operand order checked by reading, not run on hardware
//
#include <hip/hip_runtime.h>
#include <hip/hip_bf16.h>

// ---------------------------------------------------------------------------
// PointNet++ part-seg forward for gfx1250 (MI455X).
// Activations: f16, (B, M, Cpad) layout, Cpad % 32 == 0, pads zeroed.
// Weights: packed to f16 (Cout, Cpad) before each GEMM.
// GEMM: each wave computes a 32(out-ch) x 16(col) tile -> per 32-K chunk:
//   6x global_load_b128 + 2x v_wmma_f32_16x16x32_f16 (B operand reused).
// BN runs on the fp32 GEMM output; apply-kernel repacks to f16 for next layer.
// Deterministic: no atomics, fixed geometry, every buffer fully rewritten.
// ---------------------------------------------------------------------------

typedef __attribute__((ext_vector_type(16))) _Float16 v16h;
typedef __attribute__((ext_vector_type(8)))  _Float16 v8h;
typedef __attribute__((ext_vector_type(8)))  float    v8f;

constexpr int BATCH = 32;
constexpr int NPTS  = 1024;
constexpr int NCLS_ = 40;
constexpr int UPF   = 4;
constexpr int NSAMP = 32;

// ---------------------------------------------------------------------------
// Weight pack: W (Cout, Cin) fp32 -> Wh (Cout, Cin_pad) f16, zero-padded.
// ---------------------------------------------------------------------------
__global__ void pack_weights_kernel(const float* __restrict__ W, _Float16* __restrict__ Wh,
                                    int Cin, int Cin_pad, int Cout)
{
    const int t = blockIdx.x * blockDim.x + threadIdx.x;
    if (t >= Cout * Cin_pad) return;
    const int o = t / Cin_pad, k = t % Cin_pad;
    Wh[t] = (k < Cin) ? (_Float16)W[(size_t)o * Cin + k] : (_Float16)0.f;
}

// ---------------------------------------------------------------------------
// WMMA GEMM: Y[b,o,m] = sum_k Wh[o,k] * Xh[b,m,k] + bias[o]  (optional relu)
// Xh: (B, M, Cin_pad) f16, Wh: (Cout, Cin_pad) f16, Y: (B, Cout, M) fp32.
// Requires: Cin_pad % 32 == 0, M % 128 == 0.
// One wave -> 32 output channels x 16 columns (2 WMMA tiles, shared B operand).
// grid = (B, M/128, ceil(Cout/32)), block = 256 (8 waves).
// ---------------------------------------------------------------------------
__global__ void gemm_f16_kernel(const _Float16* __restrict__ Xh,
                                const _Float16* __restrict__ Wh,
                                const float* __restrict__ bias,
                                float* __restrict__ Y,
                                int Cin_pad, int Cout, int M, int relu)
{
    const int lane  = threadIdx.x & 31;
    const int wave  = threadIdx.x >> 5;
    const int b     = blockIdx.x;
    const int m0    = (blockIdx.y * 8 + wave) * 16;
    const int o0    = blockIdx.z * 32;
    const int col   = lane & 15;
    const int khalf = (lane >> 4) * 8;     // lanes 16-31 hold the +8 K-halves

    int orow0 = o0 + col;       if (orow0 >= Cout) orow0 = Cout - 1;
    int orow1 = o0 + 16 + col;  if (orow1 >= Cout) orow1 = Cout - 1;
    const _Float16* Wr0 = Wh + (size_t)orow0 * Cin_pad;
    const _Float16* Wr1 = Wh + (size_t)orow1 * Cin_pad;
    const _Float16* Xr  = Xh + ((size_t)b * M + (m0 + col)) * Cin_pad;

    v8f acc0 = {}, acc1 = {};
    for (int k0 = 0; k0 < Cin_pad; k0 += 32) {
        const v8h x0 = *(const v8h*)(Xr  + k0 + khalf);
        const v8h x1 = *(const v8h*)(Xr  + k0 + 16 + khalf);
        const v8h a0 = *(const v8h*)(Wr0 + k0 + khalf);
        const v8h a1 = *(const v8h*)(Wr0 + k0 + 16 + khalf);
        const v8h c0 = *(const v8h*)(Wr1 + k0 + khalf);
        const v8h c1 = *(const v8h*)(Wr1 + k0 + 16 + khalf);
        v16h bm, am0, am1;
#pragma unroll
        for (int i = 0; i < 8; ++i) {
            bm[i]  = x0[i]; bm[8 + i]  = x1[i];
            am0[i] = a0[i]; am0[8 + i] = a1[i];
            am1[i] = c0[i]; am1[8 + i] = c1[i];
        }
        acc0 = __builtin_amdgcn_wmma_f32_16x16x32_f16(
                   false, am0, false, bm, (short)0, acc0, false, false);
        acc1 = __builtin_amdgcn_wmma_f32_16x16x32_f16(
                   false, am1, false, bm, (short)0, acc1, false, false);
    }

    // D layout: VGPR v -> row (v + 8*(lane>=16)), col = lane%16
    const int rbase = (lane >> 4) * 8;
    const int m     = m0 + col;
#pragma unroll
    for (int v = 0; v < 8; ++v) {
        const int o = o0 + rbase + v;
        if (o < Cout) {
            float val = acc0[v] + bias[o];
            if (relu) val = val > 0.f ? val : 0.f;
            Y[((size_t)b * Cout + o) * M + m] = val;
        }
    }
#pragma unroll
    for (int v = 0; v < 8; ++v) {
        const int o = o0 + 16 + rbase + v;
        if (o < Cout) {
            float val = acc1[v] + bias[o];
            if (relu) val = val > 0.f ? val : 0.f;
            Y[((size_t)b * Cout + o) * M + m] = val;
        }
    }
}

// ---------------------------------------------------------------------------
// BN statistics: per-channel mean/var over (batch, spatial) of Y (B, C, M).
// One block per channel, deterministic tree reduction.
// ---------------------------------------------------------------------------
__global__ void bn_stats_kernel(const float* __restrict__ Y,
                                float* __restrict__ mean, float* __restrict__ varr,
                                int C, int M)
{
    const int c = blockIdx.x;
    const size_t total = (size_t)BATCH * M;
    float s = 0.f, s2 = 0.f;
    for (size_t i = threadIdx.x; i < total; i += blockDim.x) {
        const size_t b = i / (size_t)M;
        const size_t m = i - b * (size_t)M;
        const float v = Y[((size_t)b * C + c) * M + m];
        s += v; s2 += v * v;
    }
    __shared__ float sh[256], sh2[256];
    sh[threadIdx.x] = s; sh2[threadIdx.x] = s2;
    __syncthreads();
    for (int off = 128; off > 0; off >>= 1) {
        if ((int)threadIdx.x < off) {
            sh[threadIdx.x]  += sh[threadIdx.x + off];
            sh2[threadIdx.x] += sh2[threadIdx.x + off];
        }
        __syncthreads();
    }
    if (threadIdx.x == 0) {
        const float inv = 1.f / (float)total;
        const float mu = sh[0] * inv;
        mean[c] = mu;
        varr[c] = sh2[0] * inv - mu * mu;
    }
}

// BN apply + ReLU + repack fp32 (B,C,M) -> f16 (B, M, Cdst) at channel coff.
__global__ void bn_relu_pack_kernel(const float* __restrict__ Y,
                                    const float* __restrict__ mean,
                                    const float* __restrict__ varr,
                                    const float* __restrict__ g,
                                    const float* __restrict__ be,
                                    _Float16* __restrict__ dst,
                                    int Cout, int Cdst, int coff, int M)
{
    const int t = blockIdx.x * blockDim.x + threadIdx.x;
    if (t >= BATCH * M) return;
    const int b = t / M, m = t % M;
    _Float16* o = dst + ((size_t)b * M + m) * Cdst + coff;
    for (int c = 0; c < Cout; ++c) {
        float v = (Y[((size_t)b * Cout + c) * M + m] - mean[c]) *
                  rsqrtf(varr[c] + 1e-5f) * g[c] + be[c];
        o[c] = (_Float16)(v > 0.f ? v : 0.f);
    }
}

// ReLU + repack (no BN) fp32 (B,C,M) -> f16 (B, M, Cdst) at channel coff.
__global__ void relu_pack_kernel(const float* __restrict__ Y, _Float16* __restrict__ dst,
                                 int Cout, int Cdst, int coff, int M)
{
    const int t = blockIdx.x * blockDim.x + threadIdx.x;
    if (t >= BATCH * M) return;
    const int b = t / M, m = t % M;
    _Float16* o = dst + ((size_t)b * M + m) * Cdst + coff;
    for (int c = 0; c < Cout; ++c) {
        const float v = Y[((size_t)b * Cout + c) * M + m];
        o[c] = (_Float16)(v > 0.f ? v : 0.f);
    }
}

// ---------------------------------------------------------------------------
// Farthest point sampling (one block per batch, dist in LDS).
// ---------------------------------------------------------------------------
__global__ void fps_kernel(const float* __restrict__ xyz, int n, int npoint,
                           int* __restrict__ idxout)
{
    const int b = blockIdx.x;
    const float* p = xyz + (size_t)b * n * 3;
    __shared__ float dist[1024];
    __shared__ float redv[256];
    __shared__ int   redi[256];
    __shared__ int   cur;
    for (int i = threadIdx.x; i < n; i += blockDim.x) dist[i] = 1e10f;
    if (threadIdx.x == 0) cur = 0;
    __syncthreads();
    for (int it = 0; it < npoint; ++it) {
        const int far = cur;
        if (threadIdx.x == 0) idxout[b * npoint + it] = far;
        const float cx = p[far * 3 + 0], cy = p[far * 3 + 1], cz = p[far * 3 + 2];
        float best = -1.f; int bi = 0;
        for (int i = threadIdx.x; i < n; i += blockDim.x) {
            const float dx = p[i * 3 + 0] - cx;
            const float dy = p[i * 3 + 1] - cy;
            const float dz = p[i * 3 + 2] - cz;
            const float nd = fminf(dist[i], dx * dx + dy * dy + dz * dz);
            dist[i] = nd;
            if (nd > best) { best = nd; bi = i; }
        }
        redv[threadIdx.x] = best; redi[threadIdx.x] = bi;
        __syncthreads();
        for (int off = 128; off > 0; off >>= 1) {
            if ((int)threadIdx.x < off) {
                const float ov = redv[threadIdx.x + off];
                const int   oi = redi[threadIdx.x + off];
                if (ov > redv[threadIdx.x] ||
                    (ov == redv[threadIdx.x] && oi < redi[threadIdx.x])) {
                    redv[threadIdx.x] = ov; redi[threadIdx.x] = oi;
                }
            }
            __syncthreads();
        }
        if (threadIdx.x == 0) cur = redi[0];
        __syncthreads();
    }
}

__global__ void gather_xyz_kernel(const float* __restrict__ xyz,
                                  const int* __restrict__ idx,
                                  float* __restrict__ out, int n, int npoint)
{
    const int t = blockIdx.x * blockDim.x + threadIdx.x;
    if (t >= BATCH * npoint) return;
    const int b = t / npoint;
    const int k = idx[t];
    out[(size_t)t * 3 + 0] = xyz[((size_t)b * n + k) * 3 + 0];
    out[(size_t)t * 3 + 1] = xyz[((size_t)b * n + k) * 3 + 1];
    out[(size_t)t * 3 + 2] = xyz[((size_t)b * n + k) * 3 + 2];
}

__global__ void ball_query_kernel(const float* __restrict__ xyz,
                                  const float* __restrict__ new_xyz,
                                  int n, int npoint, float r2,
                                  int* __restrict__ idx)
{
    const int t = blockIdx.x * blockDim.x + threadIdx.x;
    if (t >= BATCH * npoint) return;
    const int b = t / npoint;
    const float* p = xyz + (size_t)b * n * 3;
    const float qx = new_xyz[(size_t)t * 3 + 0];
    const float qy = new_xyz[(size_t)t * 3 + 1];
    const float qz = new_xyz[(size_t)t * 3 + 2];
    int buf[NSAMP];
    int cnt = 0;
    for (int i = 0; i < n && cnt < NSAMP; ++i) {
        const float dx = p[i * 3 + 0] - qx;
        const float dy = p[i * 3 + 1] - qy;
        const float dz = p[i * 3 + 2] - qz;
        if (dx * dx + dy * dy + dz * dz <= r2) buf[cnt++] = i;
    }
    if (cnt == 0) { buf[0] = 0; cnt = 1; }
    for (int s = 0; s < NSAMP; ++s)
        idx[(size_t)t * NSAMP + s] = (s < cnt) ? buf[s] : buf[0];
}

// ---------------------------------------------------------------------------
// Grouping: Gh (B, NSAMP*npoint, Cin_pad) f16, m = s*npoint + j.
// c<3: centered xyz; 3..3+Cprev: gathered feats; rest zero pad.
// ---------------------------------------------------------------------------
__global__ void group_pack_kernel(const float* __restrict__ xyz,
                                  const float* __restrict__ new_xyz,
                                  const _Float16* __restrict__ feat,
                                  const int* __restrict__ idx,
                                  _Float16* __restrict__ Gh,
                                  int n, int npoint, int Cprev, int Cin_pad)
{
    const size_t total = (size_t)BATCH * NSAMP * npoint;
    const size_t t = (size_t)blockIdx.x * blockDim.x + threadIdx.x;
    if (t >= total) return;
    const int b   = (int)(t / ((size_t)NSAMP * npoint));
    const int rem = (int)(t % ((size_t)NSAMP * npoint));
    const int s = rem / npoint;
    const int j = rem % npoint;
    const int k = idx[((size_t)b * npoint + j) * NSAMP + s];
    _Float16* o = Gh + ((size_t)b * NSAMP * npoint + (size_t)s * npoint + j) * Cin_pad;
#pragma unroll
    for (int c = 0; c < 3; ++c)
        o[c] = (_Float16)(xyz[((size_t)b * n + k) * 3 + c] -
                          new_xyz[((size_t)b * npoint + j) * 3 + c]);
    const _Float16* f = feat + ((size_t)b * n + k) * Cprev;   // unused if Cprev==0
    for (int c = 0; c < Cprev; ++c) o[3 + c] = f[c];
    for (int c = 3 + Cprev; c < Cin_pad; ++c) o[c] = (_Float16)0.f;
}

// Max over samples: out (B, npoint, C) from Xh (B, NSAMP*npoint, C), m=s*np+j.
__global__ void maxpool_f16_kernel(const _Float16* __restrict__ Xh,
                                   _Float16* __restrict__ out, int C, int npoint)
{
    const size_t total = (size_t)BATCH * npoint * C;
    const size_t t = (size_t)blockIdx.x * blockDim.x + threadIdx.x;
    if (t >= total) return;
    const int c = (int)(t % C);
    const int j = (int)((t / C) % npoint);
    const int b = (int)(t / ((size_t)C * npoint));
    const size_t M = (size_t)NSAMP * npoint;
    float mx = -3.4e38f;
    for (int s = 0; s < NSAMP; ++s)
        mx = fmaxf(mx, (float)Xh[((size_t)b * M + (size_t)s * npoint + j) * C + c]);
    out[((size_t)b * npoint + j) * C + c] = (_Float16)mx;
}

// 3-NN inverse-distance interp: out f16 (B, n1, C) from p2 f16 (B, n2, C).
__global__ void three_interp_kernel(const float* __restrict__ xyz1,
                                    const float* __restrict__ xyz2,
                                    const _Float16* __restrict__ p2,
                                    _Float16* __restrict__ out,
                                    int n1, int n2, int C)
{
    const int t = blockIdx.x * blockDim.x + threadIdx.x;
    if (t >= BATCH * n1) return;
    const int b = t / n1, i = t % n1;
    const float* q = xyz1 + ((size_t)b * n1 + i) * 3;
    const float* p = xyz2 + (size_t)b * n2 * 3;
    float d0 = 1e30f, d1 = 1e30f, d2 = 1e30f;
    int i0 = 0, i1 = 0, i2 = 0;
    for (int k = 0; k < n2; ++k) {
        const float dx = p[k * 3 + 0] - q[0];
        const float dy = p[k * 3 + 1] - q[1];
        const float dz = p[k * 3 + 2] - q[2];
        const float d = dx * dx + dy * dy + dz * dz;
        if (d < d0)      { d2 = d1; i2 = i1; d1 = d0; i1 = i0; d0 = d; i0 = k; }
        else if (d < d1) { d2 = d1; i2 = i1; d1 = d;  i1 = k; }
        else if (d < d2) { d2 = d;  i2 = k; }
    }
    float w0 = 1.f / (d0 + 1e-8f), w1 = 1.f / (d1 + 1e-8f), w2 = 1.f / (d2 + 1e-8f);
    const float ws = w0 + w1 + w2;
    w0 /= ws; w1 /= ws; w2 /= ws;
    const _Float16* pb = p2 + (size_t)b * n2 * C;
    _Float16* ob = out + ((size_t)b * n1 + i) * C;
    for (int c = 0; c < C; ++c) {
        ob[c] = (_Float16)(w0 * (float)pb[(size_t)i0 * C + c] +
                           w1 * (float)pb[(size_t)i1 * C + c] +
                           w2 * (float)pb[(size_t)i2 * C + c]);
    }
}

// Channel-concat with optional spatial tiling (Mdst multiple of Msrc).
__global__ void copy_chan_kernel(const _Float16* __restrict__ src, _Float16* __restrict__ dst,
                                 int Csrc, int Cdst, int coff, int Msrc, int Mdst)
{
    const size_t total = (size_t)BATCH * Mdst * Csrc;
    const size_t t = (size_t)blockIdx.x * blockDim.x + threadIdx.x;
    if (t >= total) return;
    const int c = (int)(t % Csrc);
    const int m = (int)((t / Csrc) % Mdst);
    const int b = (int)(t / ((size_t)Csrc * Mdst));
    dst[((size_t)b * Mdst + m) * Cdst + coff + c] =
        src[((size_t)b * Msrc + (m % Msrc)) * Csrc + c];
}

// Broadcast one-hot labels into channels [coff, coff+NCLS_).
__global__ void labels_chan_kernel(const float* __restrict__ labels, _Float16* __restrict__ dst,
                                   int Cdst, int coff, int Mdst)
{
    const size_t total = (size_t)BATCH * Mdst * NCLS_;
    const size_t t = (size_t)blockIdx.x * blockDim.x + threadIdx.x;
    if (t >= total) return;
    const int c = (int)(t % NCLS_);
    const int m = (int)((t / NCLS_) % Mdst);
    const int b = (int)(t / ((size_t)NCLS_ * Mdst));
    dst[((size_t)b * Mdst + m) * Cdst + coff + c] = (_Float16)labels[(size_t)b * NCLS_ + c];
}

// Raw (B, n, 3) xyz into channels [coff, coff+3).
__global__ void xyz_chan_kernel(const float* __restrict__ xyz, _Float16* __restrict__ dst,
                                int Cdst, int coff, int n)
{
    const size_t total = (size_t)BATCH * n * 3;
    const size_t t = (size_t)blockIdx.x * blockDim.x + threadIdx.x;
    if (t >= total) return;
    const int c = (int)(t % 3);
    const int m = (int)((t / 3) % n);
    const int b = (int)(t / ((size_t)3 * n));
    dst[((size_t)b * n + m) * Cdst + coff + c] = (_Float16)xyz[((size_t)b * n + m) * 3 + c];
}

// Zero pad channels [coff, coff+cnt).
__global__ void zero_chan_kernel(_Float16* __restrict__ dst, int Cdst, int coff, int cnt, int Mdst)
{
    const size_t total = (size_t)BATCH * Mdst * cnt;
    const size_t t = (size_t)blockIdx.x * blockDim.x + threadIdx.x;
    if (t >= total) return;
    const int c = (int)(t % cnt);
    const int m = (int)((t / cnt) % Mdst);
    const int b = (int)(t / ((size_t)cnt * Mdst));
    dst[((size_t)b * Mdst + m) * Cdst + coff + c] = (_Float16)0.f;
}

// ---------------------------------------------------------------------------
// Host side
// ---------------------------------------------------------------------------
static inline unsigned cdivu(size_t a, size_t b) { return (unsigned)((a + b - 1) / b); }
static inline int pad32(int c) { return (c + 31) & ~31; }

static void run_gemm(const _Float16* Xh, const float* Wsrc, const float* bias,
                     _Float16* wpack, float* Y, int Cin, int Cin_pad,
                     int Cout, int M, int relu, hipStream_t s)
{
    pack_weights_kernel<<<cdivu((size_t)Cout * Cin_pad, 256), 256, 0, s>>>(
        Wsrc, wpack, Cin, Cin_pad, Cout);
    dim3 grid((unsigned)BATCH, (unsigned)(M / 128), cdivu(Cout, 32));
    gemm_f16_kernel<<<grid, 256, 0, s>>>(Xh, wpack, bias, Y, Cin_pad, Cout, M, relu);
}

extern "C" void kernel_launch(void* const* d_in, const int* in_sizes, int n_in,
                              void* d_out, int out_size, void* d_ws, size_t ws_size,
                              hipStream_t stream)
{
    (void)in_sizes; (void)n_in; (void)out_size; (void)ws_size;
    const float* pc     = (const float*)d_in[0];   // (B, N, 3)
    const float* labels = (const float*)d_in[1];   // (B, NCLS)
    auto P = [&](int i) { return (const float*)d_in[i]; };

    char*  base = (char*)d_ws;
    size_t off  = 0;
    auto allocb = [&](size_t bytes) -> char* {
        char* p = base + off;
        off += (bytes + 255) & ~(size_t)255;
        return p;
    };
    auto allocf = [&](size_t n) { return (float*)allocb(n * 4); };
    auto alloch = [&](size_t n) { return (_Float16*)allocb(n * 2); };
    auto alloci = [&](size_t n) { return (int*)allocb(n * 4); };

    float*     l1_xyz = allocf((size_t)BATCH * 1024 * 3);
    float*     l2_xyz = allocf((size_t)BATCH * 512 * 3);
    float*     l3_xyz = allocf((size_t)BATCH * 256 * 3);
    float*     l4_xyz = allocf((size_t)BATCH * 128 * 3);
    _Float16*  l1p    = alloch((size_t)BATCH * 1024 * 64);
    _Float16*  l2p    = alloch((size_t)BATCH * 512 * 128);
    _Float16*  l3p    = alloch((size_t)BATCH * 256 * 256);
    _Float16*  l4p    = alloch((size_t)BATCH * 128 * 512);
    int*       fpsi   = alloci((size_t)BATCH * 1024);
    int*       balli  = alloci((size_t)BATCH * 1024 * NSAMP);
    float*     mean   = allocf(512);
    float*     var    = allocf(512);
    _Float16*  wpack  = alloch(512 * 288);                       // max packed weight
    _Float16*  XA     = alloch((size_t)BATCH * 32768 * 64);      // 67.1M halves
    _Float16*  XB     = alloch((size_t)BATCH * 32768 * 64);
    float*     Yf     = allocf((size_t)BATCH * 64 * 32768);      // max fp32 GEMM out
    _Float16*  catb   = alloch((size_t)BATCH * 1024 * 384);      // 379 -> pad 384
    _Float16*  f2b    = alloch((size_t)BATCH * 1024 * 320);      // 296 -> pad 320
    _Float16*  npf    = alloch((size_t)BATCH * 1024 * 128);
    _Float16*  netb   = alloch((size_t)BATCH * 4096 * 192);      // 168 -> pad 192

    auto conv_bn = [&](const _Float16* X, _Float16* dst, int pbase,
                       int Cin, int Cout, int M, int Cdst, int coff) {
        run_gemm(X, P(pbase), P(pbase + 1), wpack, Yf, Cin, pad32(Cin), Cout, M, 0, stream);
        bn_stats_kernel<<<Cout, 256, 0, stream>>>(Yf, mean, var, Cout, M);
        bn_relu_pack_kernel<<<cdivu((size_t)BATCH * M, 256), 256, 0, stream>>>(
            Yf, mean, var, P(pbase + 2), P(pbase + 3), dst, Cout, Cdst, coff, M);
    };

    // ---- set-abstraction stages -------------------------------------------
    auto sa_stage = [&](const float* xyz_in, const _Float16* feat_in, int n, int np,
                        float radius, int Cprev, int pbase, int C0, int C1, int C2,
                        float* xyz_out, _Float16* points_out) {
        fps_kernel<<<BATCH, 256, 0, stream>>>(xyz_in, n, np, fpsi);
        gather_xyz_kernel<<<cdivu((size_t)BATCH * np, 256), 256, 0, stream>>>(
            xyz_in, fpsi, xyz_out, n, np);
        ball_query_kernel<<<cdivu((size_t)BATCH * np, 256), 256, 0, stream>>>(
            xyz_in, xyz_out, n, np, radius * radius, balli);
        const int Cin = 3 + Cprev;
        const int M = NSAMP * np;
        group_pack_kernel<<<cdivu((size_t)BATCH * M, 256), 256, 0, stream>>>(
            xyz_in, xyz_out, feat_in, balli, XA, n, np, Cprev, pad32(Cin));
        conv_bn(XA, XB, pbase,     Cin, C0, M, pad32(C0), 0);
        conv_bn(XB, XA, pbase + 4, C0,  C1, M, pad32(C1), 0);
        conv_bn(XA, XB, pbase + 8, C1,  C2, M, pad32(C2), 0);
        maxpool_f16_kernel<<<cdivu((size_t)BATCH * np * C2, 256), 256, 0, stream>>>(
            XB, points_out, C2, np);
    };

    sa_stage(pc,     nullptr, 1024, 1024, 0.05f, 0,   2,  32,  32,  64,  l1_xyz, l1p);
    sa_stage(l1_xyz, l1p,     1024, 512,  0.01f, 64,  14, 64,  64,  128, l2_xyz, l2p);
    sa_stage(l2_xyz, l2p,     512,  256,  0.2f,  128, 26, 128, 128, 256, l3_xyz, l3p);
    sa_stage(l3_xyz, l3p,     256,  128,  0.3f,  256, 38, 256, 256, 512, l4_xyz, l4p);

    // ---- feature propagation into catb (B, 1024, 384) ---------------------
    auto fp_stage = [&](const float* xyz2, const _Float16* p2, int n2, int C2in,
                        int pbase, int coff) {
        three_interp_kernel<<<cdivu((size_t)BATCH * NPTS, 256), 256, 0, stream>>>(
            pc, xyz2, p2, XA, NPTS, n2, C2in);
        conv_bn(XA, catb, pbase, C2in, 64, NPTS, 384, coff);
        labels_chan_kernel<<<cdivu((size_t)BATCH * NPTS * NCLS_, 256), 256, 0, stream>>>(
            labels, catb, 384, coff + 64, NPTS);
    };

    fp_stage(l4_xyz, l4p, 128, 512, 50, 0);    // up4 + labels
    fp_stage(l3_xyz, l3p, 256, 256, 54, 104);  // up3 + labels
    fp_stage(l2_xyz, l2p, 512, 128, 58, 208);  // up2 + labels

    copy_chan_kernel<<<cdivu((size_t)BATCH * NPTS * 64, 256), 256, 0, stream>>>(
        l1p, catb, 64, 384, 312, NPTS, NPTS);                       // l1_points
    xyz_chan_kernel<<<cdivu((size_t)BATCH * NPTS * 3, 256), 256, 0, stream>>>(
        pc, catb, 384, 376, NPTS);                                  // l0 xyz
    zero_chan_kernel<<<cdivu((size_t)BATCH * NPTS * 5, 256), 256, 0, stream>>>(
        catb, 384, 379, 5, NPTS);                                   // pad

    // ---- segmentation head ------------------------------------------------
    // f = relu(c2(concat)) : 379 -> 256
    run_gemm(catb, P(62), P(63), wpack, Yf, 379, 384, 256, NPTS, 0, stream);
    relu_pack_kernel<<<cdivu((size_t)BATCH * NPTS, 256), 256, 0, stream>>>(
        Yf, f2b, 256, 320, 0, NPTS);
    labels_chan_kernel<<<cdivu((size_t)BATCH * NPTS * NCLS_, 256), 256, 0, stream>>>(
        labels, f2b, 320, 256, NPTS);
    zero_chan_kernel<<<cdivu((size_t)BATCH * NPTS * 24, 256), 256, 0, stream>>>(
        f2b, 320, 296, 24, NPTS);

    // np_feat = relu(c3(f2)) : 296 -> 128
    run_gemm(f2b, P(64), P(65), wpack, Yf, 296, 320, 128, NPTS, 0, stream);
    relu_pack_kernel<<<cdivu((size_t)BATCH * NPTS, 256), 256, 0, stream>>>(
        Yf, npf, 128, 128, 0, NPTS);

    // net = [tile(np_feat, UP), lbl_up] : 168 channels, M = 4096
    const int M4 = UPF * NPTS;
    copy_chan_kernel<<<cdivu((size_t)BATCH * M4 * 128, 256), 256, 0, stream>>>(
        npf, netb, 128, 192, 0, NPTS, M4);
    labels_chan_kernel<<<cdivu((size_t)BATCH * M4 * NCLS_, 256), 256, 0, stream>>>(
        labels, netb, 192, 128, M4);
    zero_chan_kernel<<<cdivu((size_t)BATCH * M4 * 24, 256), 256, 0, stream>>>(
        netb, 192, 168, 24, M4);

    // coord = relu(c4(net)) : 168 -> 64
    run_gemm(netb, P(66), P(67), wpack, Yf, 168, 192, 64, M4, 0, stream);
    relu_pack_kernel<<<cdivu((size_t)BATCH * M4, 256), 256, 0, stream>>>(
        Yf, XA, 64, 64, 0, M4);

    // out = relu(c5(coord)) : 64 -> 3 -> d_out (B, 3, 4096) fp32
    run_gemm(XA, P(68), P(69), wpack, (float*)d_out, 64, 64, 3, M4, 1, stream);
}
